// MSAModule_2061584302844
// MI455X (gfx1250) — matrix operand order repaired and checked by reading, not compiled, not run on hardware
//
#include <hip/hip_runtime.h>
#include <string.h>

// ============================================================================
// MSA module (AF3-style) for gfx1250. All matrix contractions go through
// v_wmma_f32_16x16x32_bf16 (bf16 A/B, fp32 accum). LN/softmax/activations fp32.
//
// Input order assumption (harness flattens setup_inputs() dict in insertion
// order, params pytree in JAX order = dict keys sorted alphabetically):
//   0: msa [1,256,192,64] f32     1: single_repr [1,192,384] f32
//   2: pairwise_repr [1,192,192,128] f32
//   3: mask [1,192] bool(u8)      4: msa_mask [1,256] bool(u8)
//   5..: params leaves:
//     layers[0], layers[1] each (keys sorted):
//       mt : ln_b, ln_s, w1, w2
//       opm: b_out, ln_b, ln_s, w_h, w_out
//       pt : ln_b, ln_s, w1, w2
//       pwa: ln_b, ln_s, lnp_b, lnp_s, w_b, w_out, w_vg
//       tae: b_g, bq, ln_b, ln_s, w_bias, w_g, w_out, wkv, wq
//       tas: (same 9)
//       tmi: ln_b, ln_s, on_b, on_s, w_g, w_lr, w_out
//       tmo: (same 7)
//     then: layerscale, w_msa_init, w_single
//
// Workspace requirement: ~146 MiB (see layout in kernel_launch).
// ============================================================================

typedef unsigned short u16;

typedef __attribute__((ext_vector_type(16))) __bf16 bf16x16;
typedef __attribute__((ext_vector_type(8)))  float  f32x8;

struct AMap { long hi, lo; int div; };   // idx -> (idx/div)*hi + (idx%div)*lo

__host__ __device__ __forceinline__ long amap(AMap m, int i) {
  return (long)(i / m.div) * m.hi + (long)(i % m.div) * m.lo;
}

static AMap FLAT(long s) { AMap m; m.hi = 0; m.lo = s; m.div = 1 << 30; return m; }
static AMap MAP(long hi, long lo, int div) { AMap m; m.hi = hi; m.lo = lo; m.div = div; return m; }

__device__ __forceinline__ unsigned short f2bf(float x) {
  union { float f; unsigned u; } c; c.f = x;
  unsigned r = c.u + 0x7fffu + ((c.u >> 16) & 1u);
  return (unsigned short)(r >> 16);
}
__device__ __forceinline__ float bf2f(unsigned short h) {
  union { float f; unsigned u; } c; c.u = ((unsigned)h) << 16;
  return c.f;
}
__device__ __forceinline__ float ld_any(const void* p, long idx, int isBf16) {
  return isBf16 ? bf2f(((const unsigned short*)p)[idx]) : ((const float*)p)[idx];
}
__device__ __forceinline__ float actf(int a, float x) {
  if (a == 1) return 1.0f / (1.0f + __expf(-x));            // sigmoid
  if (a == 2) return x / (1.0f + __expf(-x));               // silu
  return x;
}

// ---------------------------------------------------------------------------
// Generic batched/strided WMMA GEMM.  D[m,n] over batch (z1,z2):
//   acc1 = sum_k A[m,k] * B[k, nOff1+n]
//   acc2 = sum_k A[m,k] * B[k, nOff2+n]        (if nOff2 >= 0)
//   v    = dual ? actf(act,acc2)*acc1 : (act ? actf(act, acc1*s + bias) : acc1*s)
//   v    = v * maskfactor (+ bias post if act==0); s = scale * (*scalePtr), applied to acc1 and acc2
//   v    = residMul ? v*resid : v+resid
//
// REQUIREMENTS (all call sites obey): K % 32 == 0; every k-address-map is
// linear across 32-aligned steps (div==32 or FLAT), so per-element addresses
// advance by the constant amap(map,32) each k-step (no divisions in the loop).
// ---------------------------------------------------------------------------
struct GemmP {
  const void* A; const void* B; void* C;
  const void* resid;
  const float* biasN;
  const float* scalePtr;
  const unsigned char* mask;
  AMap am, ak, bk, bn, cm, cn, rm, rn;
  long aZ1, aZ2, bZ1, bZ2, cZ1, cZ2, rZ1, rZ2;
  int M, N, K;
  int Z1, Z2;
  int nOff1, nOff2;
  int act;
  float scale;
  int maskMode, maskW, maskRowOff;   // 0 none, 1 pair mask[i]&&mask[j], 2 row mask[m/W]
  int residBf16, residMul;
  int aBf16, bBf16, outBf16;
};

__global__ __launch_bounds__(32) void k_gemm(GemmP p) {
  const int lane = threadIdx.x & 31;
  const int r  = lane & 15, hh = lane >> 4;
  const int m0 = blockIdx.x * 16, n0 = blockIdx.y * 16;
  const int z  = blockIdx.z;
  const int z1 = z / p.Z2, z2 = z % p.Z2;
  const long aZ = (long)z1 * p.aZ1 + (long)z2 * p.aZ2;
  const long bZ = (long)z1 * p.bZ1 + (long)z2 * p.bZ2;
  const long cZ = (long)z1 * p.cZ1 + (long)z2 * p.cZ2;
  const long rZ = (long)z1 * p.rZ1 + (long)z2 * p.rZ2;
  const bool dual = p.nOff2 >= 0;

  const int am_ = m0 + r;
  const int bn_ = n0 + r;
  const bool mok = am_ < p.M;
  const bool nok = bn_ < p.N;

  // ---- loop-invariant address setup (the only divisions in this kernel) ----
  long aCur  = aZ + amap(p.am, mok ? am_ : 0);
  long bCur1 = bZ + amap(p.bn, p.nOff1 + (nok ? bn_ : 0));
  long bCur2 = dual ? (bZ + amap(p.bn, p.nOff2 + (nok ? bn_ : 0))) : 0;
  int offA[16], offB[16];
#pragma unroll
  for (int e = 0; e < 16; ++e) {
    int cA = (hh << 3) + (e & 7) + ((e & 8) ? 16 : 0);   // A frag k-constant (<32)
    int cB = (hh << 3) + (e & 7) + ((e & 8) ? 16 : 0);   // B frag k-constant (<32)
    offA[e] = (int)amap(p.ak, cA);
    offB[e] = (int)amap(p.bk, cB);
  }
  const long dA = amap(p.ak, 32);       // per-step advance (linear across steps)
  const long dB = amap(p.bk, 32);

  f32x8 acc1 = {0.f,0.f,0.f,0.f,0.f,0.f,0.f,0.f};
  f32x8 acc2 = {0.f,0.f,0.f,0.f,0.f,0.f,0.f,0.f};

  const int steps = (p.K + 31) >> 5;
  for (int s = 0; s < steps; ++s) {
    union { bf16x16 v; u16 u[16]; } fa, fb;
    if (mok) {
      if (p.aBf16) {
        const u16* a16 = (const u16*)p.A;
#pragma unroll
        for (int e = 0; e < 16; ++e) fa.u[e] = a16[aCur + offA[e]];
      } else {
        const float* a32 = (const float*)p.A;
#pragma unroll
        for (int e = 0; e < 16; ++e) fa.u[e] = f2bf(a32[aCur + offA[e]]);
      }
    } else {
#pragma unroll
      for (int e = 0; e < 16; ++e) fa.u[e] = 0;
    }
    if (nok) {
      if (p.bBf16) {
        const u16* b16 = (const u16*)p.B;
#pragma unroll
        for (int e = 0; e < 16; ++e) fb.u[e] = b16[bCur1 + offB[e]];
      } else {
        const float* b32 = (const float*)p.B;
#pragma unroll
        for (int e = 0; e < 16; ++e) fb.u[e] = f2bf(b32[bCur1 + offB[e]]);
      }
    } else {
#pragma unroll
      for (int e = 0; e < 16; ++e) fb.u[e] = 0;
    }
    acc1 = __builtin_amdgcn_wmma_f32_16x16x32_bf16(false, fa.v, false, fb.v, (short)0, acc1, false, false);
    if (dual) {
      union { bf16x16 v; u16 u[16]; } fg;
      if (nok) {
        if (p.bBf16) {
          const u16* b16 = (const u16*)p.B;
#pragma unroll
          for (int e = 0; e < 16; ++e) fg.u[e] = b16[bCur2 + offB[e]];
        } else {
          const float* b32 = (const float*)p.B;
#pragma unroll
          for (int e = 0; e < 16; ++e) fg.u[e] = f2bf(b32[bCur2 + offB[e]]);
        }
      } else {
#pragma unroll
        for (int e = 0; e < 16; ++e) fg.u[e] = 0;
      }
      acc2 = __builtin_amdgcn_wmma_f32_16x16x32_bf16(false, fa.v, false, fg.v, (short)0, acc2, false, false);
      bCur2 += dB;
    }
    aCur += dA; bCur1 += dB;
  }

  // ---- epilogue ----
  const int col = n0 + (lane & 15);
  if (col >= p.N) return;
  const float sp   = p.scalePtr ? p.scalePtr[0] : 1.0f;
  const float bias = p.biasN ? p.biasN[col] : 0.0f;
  const long cn_ = amap(p.cn, col);
  const long rn_ = p.resid ? amap(p.rn, col) : 0;

#pragma unroll
  for (int v = 0; v < 8; ++v) {
    int row = m0 + v + ((lane >> 4) << 3);
    if (row >= p.M) continue;
    float v1 = acc1[v] * p.scale * sp;
    float out;
    if (dual) {
      float v2 = acc2[v] * p.scale * sp;
      out = actf(p.act, v2) * v1;
    } else if (p.act) {
      out = actf(p.act, v1 + bias);
    } else {
      out = v1;
    }
    if (p.mask) {
      float mf;
      if (p.maskMode == 1) {
        int i = row / p.maskW + p.maskRowOff;
        int j = row % p.maskW;
        mf = (p.mask[i] && p.mask[j]) ? 1.0f : 0.0f;
      } else {
        mf = p.mask[row / p.maskW] ? 1.0f : 0.0f;
      }
      out *= mf;
    }
    if (!dual && !p.act && p.biasN) out += bias;
    if (p.resid) {
      long ra = rZ + amap(p.rm, row) + rn_;
      float rv = ld_any(p.resid, ra, p.residBf16);
      out = p.residMul ? out * rv : out + rv;
    }
    long ca = cZ + amap(p.cm, row) + cn_;
    if (p.outBf16) ((unsigned short*)p.C)[ca] = f2bf(out);
    else           ((float*)p.C)[ca] = out;
  }
}

// ---------------------------------------------------------------------------
// LayerNorm: one wave per token.
// ---------------------------------------------------------------------------
__global__ void k_layernorm(const float* __restrict__ x, const float* __restrict__ s,
                            const float* __restrict__ b, float* __restrict__ y,
                            int T, int D) {
  int t = blockIdx.x * (blockDim.x >> 5) + (threadIdx.x >> 5);
  int lane = threadIdx.x & 31;
  if (t >= T) return;
  const float* xr = x + (long)t * D;
  float sum = 0.f, sq = 0.f;
  for (int i = lane; i < D; i += 32) { float v = xr[i]; sum += v; sq += v * v; }
  for (int m = 16; m; m >>= 1) { sum += __shfl_xor(sum, m, 32); sq += __shfl_xor(sq, m, 32); }
  float mu  = sum / (float)D;
  float var = sq / (float)D - mu * mu;
  float inv = rsqrtf(fmaxf(var, 0.f) + 1e-5f);
  float* yr = y + (long)t * D;
  for (int i = lane; i < D; i += 32) yr[i] = (xr[i] - mu) * inv * s[i] + b[i];
}

// ---------------------------------------------------------------------------
// Masked softmax over L elements per row; strided addressing; bf16 capable.
// ---------------------------------------------------------------------------
struct SmP {
  const void* x; void* y; const unsigned char* mask;
  AMap rmap; long cs;
  int R, L, inBf16, outBf16;
};

__global__ void k_softmax(SmP p) {
  int r = blockIdx.x * (blockDim.x >> 5) + (threadIdx.x >> 5);
  int lane = threadIdx.x & 31;
  if (r >= p.R) return;
  long base = amap(p.rmap, r);
  float mx = -3.0e38f;
  for (int j = lane; j < p.L; j += 32) {
    if (p.mask && !p.mask[j]) continue;
    mx = fmaxf(mx, ld_any(p.x, base + (long)j * p.cs, p.inBf16));
  }
  for (int m = 16; m; m >>= 1) mx = fmaxf(mx, __shfl_xor(mx, m, 32));
  float sum = 0.f;
  for (int j = lane; j < p.L; j += 32) {
    if (!p.mask || p.mask[j]) sum += __expf(ld_any(p.x, base + (long)j * p.cs, p.inBf16) - mx);
  }
  for (int m = 16; m; m >>= 1) sum += __shfl_xor(sum, m, 32);
  float inv = 1.0f / fmaxf(sum, 1e-30f);
  for (int j = lane; j < p.L; j += 32) {
    long a = base + (long)j * p.cs;
    float e = 0.f;
    if (!p.mask || p.mask[j]) e = __expf(ld_any(p.x, a, p.inBf16) - mx) * inv;
    if (p.outBf16) ((unsigned short*)p.y)[a] = f2bf(e);
    else           ((float*)p.y)[a] = e;
  }
}

__global__ void k_ew_mul(float* y, const float* g, long n) {
  long i = (long)blockIdx.x * blockDim.x + threadIdx.x;
  long st = (long)gridDim.x * blockDim.x;
  for (; i < n; i += st) y[i] *= g[i];
}

__global__ void k_scale_out(const float* pw, const float* ls, float* out, long n, int D) {
  long i = (long)blockIdx.x * blockDim.x + threadIdx.x;
  long st = (long)gridDim.x * blockDim.x;
  for (; i < n; i += st) out[i] = pw[i] * ls[i % D];
}

__global__ void k_invnum(const unsigned char* mm, int S, float* out) {
  __shared__ float sh[8];
  int t = threadIdx.x;
  float v = (t < S) ? (mm[t] ? 1.f : 0.f) : 0.f;
  for (int m = 16; m; m >>= 1) v += __shfl_xor(v, m, 32);
  if ((t & 31) == 0) sh[t >> 5] = v;
  __syncthreads();
  if (t == 0) {
    float s = 0.f;
    for (int i = 0; i < 8; ++i) s += sh[i];
    out[0] = 1.0f / fmaxf(s, 1e-5f);
  }
}

// ===========================================================================
// Host side
// ===========================================================================
static GemmP G(const void* A, const void* B, void* C, int M, int N, int K) {
  GemmP p; memset(&p, 0, sizeof(p));
  p.A = A; p.B = B; p.C = C; p.M = M; p.N = N; p.K = K;
  p.am = FLAT(K);  p.ak = FLAT(1);
  p.bk = FLAT(N);  p.bn = FLAT(1);
  p.cm = FLAT(N);  p.cn = FLAT(1);
  p.rm = FLAT(N);  p.rn = FLAT(1);
  p.Z1 = 1; p.Z2 = 1; p.nOff1 = 0; p.nOff2 = -1;
  p.scale = 1.0f; p.maskW = 192;
  return p;
}

static void run_gemm(hipStream_t st, const GemmP& p) {
  dim3 grid((p.M + 15) / 16, (p.N + 15) / 16, (unsigned)(p.Z1 * p.Z2));
  k_gemm<<<grid, dim3(32), 0, st>>>(p);
}

static void run_ln(hipStream_t st, const float* x, const float* s, const float* b,
                   float* y, int T, int D) {
  k_layernorm<<<(T + 7) / 8, 256, 0, st>>>(x, s, b, y, T, D);
}

static void run_softmax(hipStream_t st, const SmP& p) {
  k_softmax<<<(p.R + 7) / 8, 256, 0, st>>>(p);
}

struct MTp  { const float *ln_b, *ln_s, *w1, *w2; };
struct OPMp { const float *b_out, *ln_b, *ln_s, *w_h, *w_out; };
struct PWAp { const float *ln_b, *ln_s, *lnp_b, *lnp_s, *w_b, *w_out, *w_vg; };
struct TAp  { const float *b_g, *bq, *ln_b, *ln_s, *w_bias, *w_g, *w_out, *wkv, *wq; };
struct TMp  { const float *ln_b, *ln_s, *on_b, *on_s, *w_g, *w_lr, *w_out; };
struct LayerP { MTp mt; OPMp opm; MTp pt; PWAp pwa; TAp tae, tas; TMp tmi, tmo; };

extern "C" void kernel_launch(void* const* d_in, const int* in_sizes, int n_in,
                              void* d_out, int out_size, void* d_ws, size_t ws_size,
                              hipStream_t stream) {
  (void)in_sizes; (void)n_in; (void)out_size; (void)ws_size;
  const int SS = 256, NP = 192, TOKM = 256 * 192, TOKP = 192 * 192;

  const float* msa      = (const float*)d_in[0];
  const float* single_r = (const float*)d_in[1];
  const float* pw_in    = (const float*)d_in[2];
  const unsigned char* mask     = (const unsigned char*)d_in[3];
  const unsigned char* msa_mask = (const unsigned char*)d_in[4];

  int cur = 5;
  auto nx = [&]() { return (const float*)d_in[cur++]; };
  LayerP L[2];
  for (int l = 0; l < 2; ++l) {
    L[l].mt  = { nx(), nx(), nx(), nx() };
    L[l].opm = { nx(), nx(), nx(), nx(), nx() };
    L[l].pt  = { nx(), nx(), nx(), nx() };
    L[l].pwa = { nx(), nx(), nx(), nx(), nx(), nx(), nx() };
    L[l].tae = { nx(), nx(), nx(), nx(), nx(), nx(), nx(), nx(), nx() };
    L[l].tas = { nx(), nx(), nx(), nx(), nx(), nx(), nx(), nx(), nx() };
    L[l].tmi = { nx(), nx(), nx(), nx(), nx(), nx(), nx() };
    L[l].tmo = { nx(), nx(), nx(), nx(), nx(), nx(), nx() };
  }
  const float* layerscale = nx();
  const float* w_msa_init = nx();
  const float* w_single   = nx();

  // ---- workspace layout (float units) ----
  float* W    = (float*)d_ws;
  float* scal = W;                                // [16]
  float* Fm   = W + 16;                           // MSA act [49152,64]
  float* Fpw  = Fm + (long)TOKM * 64;             // pair act [36864,128]
  float* Fxn  = Fpw + (long)TOKP * 128;           // LN scratch [36864,128]
  float* S0   = Fxn + (long)TOKP * 128;           // scratch arena (~25.5M floats)

  k_invnum<<<1, 256, 0, stream>>>(msa_mask, SS, scal);
  hipMemcpyAsync(Fpw, pw_in, (size_t)TOKP * 128 * sizeof(float),
                 hipMemcpyDeviceToDevice, stream);

  // ---- init: m = msa @ w_msa_init + broadcast(single @ w_single) ----
  {
    float* sbuf = S0;                            // [192,64]
    GemmP p = G(single_r, w_single, sbuf, NP, 64, 384);
    run_gemm(stream, p);
    p = G(msa, w_msa_init, Fm, TOKM, 64, 64);
    p.resid = sbuf; p.rm = MAP(0, 64, 192); p.rn = FLAT(1);
    run_gemm(stream, p);
  }

  for (int l = 0; l < 2; ++l) {
    const LayerP& ly = L[l];

    // ================= OuterProductMean: pw += OPM(m) =================
    {
      run_ln(stream, Fm, ly.opm.ln_s, ly.opm.ln_b, Fxn, TOKM, 64);
      u16* AB = (u16*)S0;                                // [49152,64] bf16 (a|bb)
      u16* OP = (u16*)(S0 + 1572864);                    // [48,32,6144] bf16
      GemmP p = G(Fxn, ly.opm.w_h, AB, TOKM, 64, 64);
      p.outBf16 = 1; p.mask = msa_mask; p.maskMode = 2; p.maskW = 192;
      run_gemm(stream, p);
      for (int ic = 0; ic < 4; ++ic) {
        int i0 = ic * 48;
        // op[il][d][(j,e)] = sum_s a[s,i0+il,d] * bb[s,j,e]
        p = G((const u16*)AB + (long)i0 * 64, (const u16*)AB + 32, OP, 32, 6144, 256);
        p.aBf16 = 1; p.bBf16 = 1; p.outBf16 = 1;
        p.am = FLAT(1);      p.ak = FLAT(12288);
        p.bk = FLAT(12288);  p.bn = MAP(64, 1, 32);
        p.cm = FLAT(6144);   p.cn = FLAT(1);
        p.Z1 = 1; p.Z2 = 48; p.aZ2 = 64; p.bZ2 = 0; p.cZ2 = 196608;
        run_gemm(stream, p);
        // pw[(i0+il)*192+j, c] += pm * (op/num) @ w_out + b_out
        p = G(OP, ly.opm.w_out, Fpw + (long)i0 * 192 * 128, 48 * 192, 128, 1024);
        p.aBf16 = 1;
        p.am = MAP(196608, 32, 192); p.ak = MAP(6144, 1, 32);
        p.scalePtr = scal; p.biasN = ly.opm.b_out;
        p.mask = mask; p.maskMode = 1; p.maskW = 192; p.maskRowOff = i0;
        p.resid = Fpw + (long)i0 * 192 * 128; p.rm = FLAT(128); p.rn = FLAT(1);
        run_gemm(stream, p);
      }
    }

    // ================= MSAPairWeightedAveraging: m += PWA(m, pw) =================
    {
      run_ln(stream, Fm, ly.pwa.ln_s, ly.pwa.ln_b, Fxn, TOKM, 64);
      u16*  VB = (u16*)S0;                  // values bf16 [49152,256]
      u16*  GB = (u16*)(S0 + 6291456);      // sigmoid(gates) bf16 [49152,256]
      float* WL = S0 + 12582912;            // logits/softmax fp32 [36864,8]
      float* OB = S0 + 12877824;            // out fp32 [49152,256]
      GemmP p = G(Fxn, ly.pwa.w_vg, VB, TOKM, 256, 64);
      p.bk = FLAT(512); p.outBf16 = 1;
      run_gemm(stream, p);
      p = G(Fxn, ly.pwa.w_vg, GB, TOKM, 256, 64);
      p.bk = FLAT(512); p.nOff1 = 256; p.act = 1; p.outBf16 = 1;
      run_gemm(stream, p);
      run_ln(stream, Fpw, ly.pwa.lnp_s, ly.pwa.lnp_b, Fxn, TOKP, 128);
      p = G(Fxn, ly.pwa.w_b, WL, TOKP, 8, 128);
      run_gemm(stream, p);
      SmP sm; memset(&sm, 0, sizeof(sm));
      sm.x = WL; sm.y = WL; sm.mask = mask;
      sm.rmap = MAP(1536, 1, 8); sm.cs = 8; sm.R = 192 * 8; sm.L = 192;
      run_softmax(stream, sm);
      // out[(s,i),h*32+d] = sum_j WL[i,j,h] * VB[(s,j),h*32+d] ; * GB
      p = G(WL, VB, OB, 192, 32, 192);
      p.bBf16 = 1;
      p.am = FLAT(1536); p.ak = FLAT(8);
      p.bk = FLAT(256);  p.bn = FLAT(1);
      p.cm = FLAT(256);  p.cn = FLAT(1);
      p.Z1 = 8; p.Z2 = 256;
      p.aZ1 = 1;  p.aZ2 = 0;
      p.bZ1 = 32; p.bZ2 = 49152;
      p.cZ1 = 32; p.cZ2 = 49152;
      p.resid = GB; p.residBf16 = 1; p.residMul = 1;
      p.rm = FLAT(256); p.rn = FLAT(1); p.rZ1 = 32; p.rZ2 = 49152;
      run_gemm(stream, p);
      p = G(OB, ly.pwa.w_out, Fm, TOKM, 64, 256);
      p.resid = Fm; p.rm = FLAT(64); p.rn = FLAT(1);
      run_gemm(stream, p);
    }

    // ================= MSA transition (SwiGLU): m += T(m) =================
    {
      run_ln(stream, Fm, ly.mt.ln_s, ly.mt.ln_b, Fxn, TOKM, 64);
      float* H = S0;                        // [49152,256]
      GemmP p = G(Fxn, ly.mt.w1, H, TOKM, 256, 64);
      p.bk = FLAT(512); p.nOff2 = 256; p.act = 2;   // silu(g)*a
      run_gemm(stream, p);
      p = G(H, ly.mt.w2, Fm, TOKM, 64, 256);
      p.resid = Fm; p.rm = FLAT(64); p.rn = FLAT(1);
      run_gemm(stream, p);
    }

    // ================= Triangle multiplication (outgoing, incoming) ==========
    for (int dir = 0; dir < 2; ++dir) {
      bool outgoing = (dir == 0);
      const TMp& tm = outgoing ? ly.tmo : ly.tmi;
      run_ln(stream, Fpw, tm.ln_s, tm.ln_b, Fxn, TOKP, 128);
      u16*  LB = (u16*)S0;                  // left bf16 [36864,128]
      u16*  RB = (u16*)(S0 + 2359296);      // right bf16 [36864,128]
      float* GG = S0 + 4718592;             // sigmoid gate fp32 [36864,128]
      float* TO = S0 + 9437184;             // einsum out fp32 [36864,128]
      GemmP p = G(Fxn, tm.w_lr, LB, TOKP, 128, 128);
      p.bk = FLAT(512); p.nOff1 = 0; p.nOff2 = 256; p.act = 1; p.outBf16 = 1;
      p.mask = mask; p.maskMode = 1; p.maskW = 192;
      run_gemm(stream, p);
      p = G(Fxn, tm.w_lr, RB, TOKP, 128, 128);
      p.bk = FLAT(512); p.nOff1 = 128; p.nOff2 = 384; p.act = 1; p.outBf16 = 1;
      p.mask = mask; p.maskMode = 1; p.maskW = 192;
      run_gemm(stream, p);
      p = G(Fxn, tm.w_g, GG, TOKP, 128, 128);
      p.act = 1;
      run_gemm(stream, p);
      // batched over d (128): out[i,j,d]
      p = G(LB, RB, TO, 192, 192, 192);
      p.aBf16 = 1; p.bBf16 = 1;
      if (outgoing) {       // sum_k L[i,k,d]*R[j,k,d]
        p.am = FLAT(24576); p.ak = FLAT(128);
        p.bk = FLAT(128);   p.bn = FLAT(24576);
      } else {              // sum_k L[k,j,d]*R[k,i,d]
        p.A = RB; p.B = LB;
        p.am = FLAT(128);   p.ak = FLAT(24576);
        p.bk = FLAT(24576); p.bn = FLAT(128);
      }
      p.cm = FLAT(24576); p.cn = FLAT(128);
      p.Z1 = 1; p.Z2 = 128; p.aZ2 = 1; p.bZ2 = 1; p.cZ2 = 1;
      run_gemm(stream, p);
      run_ln(stream, TO, tm.on_s, tm.on_b, TO, TOKP, 128);       // in-place LN
      k_ew_mul<<<2048, 256, 0, stream>>>(TO, GG, (long)TOKP * 128);
      p = G(TO, tm.w_out, Fpw, TOKP, 128, 128);
      p.resid = Fpw; p.rm = FLAT(128); p.rn = FLAT(1);
      run_gemm(stream, p);
    }

    // ================= Triangle attention (starting, ending) =================
    for (int dir = 0; dir < 2; ++dir) {
      bool ending = (dir == 1);
      const TAp& ta = ending ? ly.tae : ly.tas;
      run_ln(stream, Fpw, ta.ln_s, ta.ln_b, Fxn, TOKP, 128);
      AMap amx = ending ? MAP(128, 24576, 192) : FLAT(128);   // token transpose
      u16*  QB  = (u16*)S0;                    // q bf16 [36864,128]
      u16*  KV  = (u16*)(S0 + 2359296);        // kv bf16 [36864,256]
      float* GT = S0 + 7077888;                // sigmoid(gate) fp32 [36864,128]
      float* BB = S0 + 11796480;               // bias fp32 [36864,4]
      float* OB = S0 + 11943936;               // attn out fp32 [36864,128]
      u16*  LG  = (u16*)(S0 + 16662528);       // logits bf16 chunk [48,4,192,192]
      GemmP p = G(Fxn, ta.wq, QB, TOKP, 128, 128);
      p.am = amx; p.biasN = ta.bq; p.outBf16 = 1;
      run_gemm(stream, p);
      p = G(Fxn, ta.wkv, KV, TOKP, 256, 128);
      p.am = amx; p.outBf16 = 1;
      run_gemm(stream, p);
      p = G(Fxn, ta.w_g, GT, TOKP, 128, 128);
      p.am = amx; p.biasN = ta.b_g; p.act = 1;
      run_gemm(stream, p);
      p = G(Fxn, ta.w_bias, BB, TOKP, 4, 128);
      p.am = amx;
      run_gemm(stream, p);
      for (int rc = 0; rc < 4; ++rc) {
        int r0 = rc * 48;
        // logits[rl,h,i,j] = q.k * 1/sqrt(32) + bias[i,j,h]
        p = G((const u16*)QB + (long)r0 * 24576, (const u16*)KV + (long)r0 * 49152,
              LG, 192, 192, 32);
        p.aBf16 = 1; p.bBf16 = 1; p.outBf16 = 1;
        p.am = FLAT(128); p.ak = FLAT(1);
        p.bk = FLAT(1);   p.bn = FLAT(256);
        p.cm = FLAT(192); p.cn = FLAT(1);
        p.Z1 = 48; p.Z2 = 4;
        p.aZ1 = 24576;  p.aZ2 = 32;
        p.bZ1 = 49152;  p.bZ2 = 32;
        p.cZ1 = 147456; p.cZ2 = 36864;
        p.scale = 0.17677669529663687f;
        p.resid = BB; p.rm = FLAT(768); p.rn = FLAT(4); p.rZ1 = 0; p.rZ2 = 1;
        run_gemm(stream, p);
        SmP sm; memset(&sm, 0, sizeof(sm));
        sm.x = LG; sm.y = LG; sm.mask = mask;
        sm.rmap = FLAT(192); sm.cs = 1; sm.R = 48 * 4 * 192; sm.L = 192;
        sm.inBf16 = 1; sm.outBf16 = 1;
        run_softmax(stream, sm);
        // out[rl,i,h*32+d] = sum_j attn[rl,h,i,j]*v[rl,j,h,d] ; * gate
        p = G(LG, (const u16*)KV + 128 + (long)r0 * 49152,
              OB + (long)r0 * 24576, 192, 32, 192);
        p.aBf16 = 1; p.bBf16 = 1;
        p.am = FLAT(192); p.ak = FLAT(1);
        p.bk = FLAT(256); p.bn = FLAT(1);
        p.cm = FLAT(128); p.cn = FLAT(1);
        p.Z1 = 48; p.Z2 = 4;
        p.aZ1 = 147456; p.aZ2 = 36864;
        p.bZ1 = 49152;  p.bZ2 = 32;
        p.cZ1 = 24576;  p.cZ2 = 32;
        p.resid = GT + (long)r0 * 24576; p.residMul = 1;
        p.rm = FLAT(128); p.rn = FLAT(1); p.rZ1 = 24576; p.rZ2 = 32;
        run_gemm(stream, p);
      }
      AMap cmx = ending ? MAP(128, 24576, 192) : FLAT(128);
      p = G(OB, ta.w_out, Fpw, TOKP, 128, 128);
      p.cm = cmx;
      p.resid = Fpw; p.rm = cmx; p.rn = FLAT(1);
      run_gemm(stream, p);
    }

    // ================= Pair transition (SwiGLU): pw += T(pw) =================
    {
      run_ln(stream, Fpw, ly.pt.ln_s, ly.pt.ln_b, Fxn, TOKP, 128);
      float* H = S0;                        // [36864,512]
      GemmP p = G(Fxn, ly.pt.w1, H, TOKP, 512, 128);
      p.bk = FLAT(1024); p.nOff2 = 512; p.act = 2;
      run_gemm(stream, p);
      p = G(H, ly.pt.w2, Fpw, TOKP, 128, 512);
      p.resid = Fpw; p.rm = FLAT(128); p.rn = FLAT(1);
      run_gemm(stream, p);
    }
  }

  // ---- out = pw * layerscale ----
  k_scale_out<<<2048, 256, 0, stream>>>(Fpw, layerscale, (float*)d_out,
                                        (long)TOKP * 128, 128);
}
